// MultiHeadAttention_90409061581208
// MI455X (gfx1250) — matrix operand order repaired and checked by reading, not compiled, not run on hardware
//
#include <hip/hip_runtime.h>
#include <hip/hip_bf16.h>
#include <cstdint>

#define B_  4
#define S_  2048
#define D_  1024
#define H_  16
#define DH_ 64

typedef _Float16 v8h  __attribute__((ext_vector_type(8)));
typedef _Float16 v16h __attribute__((ext_vector_type(16)));
typedef float    v8f  __attribute__((ext_vector_type(8)));
typedef unsigned int v4u  __attribute__((ext_vector_type(4)));
typedef int          v4i_ __attribute__((ext_vector_type(4)));
typedef int          v8i_ __attribute__((ext_vector_type(8)));

#if __has_builtin(__builtin_amdgcn_tensor_load_to_lds) && __has_builtin(__builtin_amdgcn_s_wait_tensorcnt)
#define ATHENA_HAVE_TDM 1
#else
#define ATHENA_HAVE_TDM 0
#endif

#if __has_builtin(__builtin_amdgcn_ds_swizzle)
#define ATHENA_HAVE_SWZ 1
#else
#define ATHENA_HAVE_SWZ 0
#endif

static __device__ inline v16h cat16(v8h lo, v8h hi) {
  return __builtin_shufflevector(lo, hi, 0,1,2,3,4,5,6,7,8,9,10,11,12,13,14,15);
}

// A fragment (16x32 f16, MxK): lane l -> row (l&15); halves 0..7 = K[(l>>4)*8 ..],
// halves 8..15 = K[16+(l>>4)*8 ..]   (ISA 7.12.2 "16-bit A-Matrix 16x32")
static __device__ inline v16h load_frag_a(const _Float16* base, int stride_h, int lane) {
  const int row = lane & 15, hi = lane >> 4;
  const _Float16* p = base + row * stride_h + hi * 8;
  v8h lo = *(const v8h*)p;
  v8h hh = *(const v8h*)(p + 16);
  return cat16(lo, hh);
}

// B fragment (32x16 f16, KxN): lane l -> column (l&15); halves 0..7 = K[(l>>4)*8 ..],
// halves 8..15 = K[16+(l>>4)*8 ..]
static __device__ inline v16h load_frag_b(const _Float16* base, int stride_h, int lane) {
  const int col = lane & 15, hi = lane >> 4;
  const _Float16* p = base + col * stride_h + hi * 8;
  v8h lo = *(const v8h*)p;
  v8h hh = *(const v8h*)(p + 16);
  return cat16(lo, hh);
}

static __device__ inline v8f wmma_f16(v16h a, v16h b, v8f c) {
  return __builtin_amdgcn_wmma_f32_16x16x32_f16(false, a, false, b, (short)0, c, false, false);
}

// 16-lane xor reductions (lanes 0-15 and 16-31 independently).
// ds_swizzle group-of-32: offset = xor_mask<<10 | or_mask<<5 | and_mask(0x1F).
#if ATHENA_HAVE_SWZ
#define ATHENA_SWZ(x, imm) __int_as_float(__builtin_amdgcn_ds_swizzle(__float_as_int(x), imm))
static __device__ inline float red16_max(float v) {
  v = fmaxf(v, ATHENA_SWZ(v, 0x041F));
  v = fmaxf(v, ATHENA_SWZ(v, 0x081F));
  v = fmaxf(v, ATHENA_SWZ(v, 0x101F));
  v = fmaxf(v, ATHENA_SWZ(v, 0x201F));
  return v;
}
static __device__ inline float red16_sum(float v) {
  v += ATHENA_SWZ(v, 0x041F);
  v += ATHENA_SWZ(v, 0x081F);
  v += ATHENA_SWZ(v, 0x101F);
  v += ATHENA_SWZ(v, 0x201F);
  return v;
}
#else
static __device__ inline float red16_max(float v) {
  for (int off = 8; off > 0; off >>= 1) v = fmaxf(v, __shfl_xor(v, off, 16));
  return v;
}
static __device__ inline float red16_sum(float v) {
  for (int off = 8; off > 0; off >>= 1) v += __shfl_xor(v, off, 16);
  return v;
}
#endif

#if ATHENA_HAVE_TDM
// TDM: DMA a 2D f16 tile (tile_d0 contiguous elems x tile_d1 rows, row stride
// row_stride_elems) from global to LDS, padding pad_amount+1 DWORDs every
// 2^(pad_interval+3) bytes of LDS destination (ISA 8.3/8.4 D# layout).
static __device__ inline void tdm_load_2d_f16(uint32_t lds_addr, const void* gaddr,
                                              uint32_t tile_d0, uint32_t tile_d1,
                                              uint32_t tensor_d0, uint32_t tensor_d1,
                                              uint64_t row_stride_elems,
                                              uint32_t pad_interval, uint32_t pad_amount,
                                              uint32_t pad_en) {
  const uint64_t ga = (uint64_t)(uintptr_t)gaddr;
  v4u g0;
  g0[0] = 1u;                                                    // count=1, user desc
  g0[1] = lds_addr;                                              // lds_addr (bytes)
  g0[2] = (uint32_t)ga;                                          // global_addr[31:0]
  g0[3] = (uint32_t)((ga >> 32) & 0x01FFFFFFu) | (2u << 30);     // ga[56:32] | type=2
  v8i_ g1;
  g1[0] = (int)((1u << 16) |                                     // data_size = 2 bytes
                (pad_en << 20) | (pad_interval << 22) | (pad_amount << 25));
  g1[1] = (int)((tensor_d0 & 0xFFFFu) << 16);                    // tensor_dim0[15:0]
  g1[2] = (int)(((tensor_d0 >> 16) & 0xFFFFu) |
                ((tensor_d1 & 0xFFFFu) << 16));
  g1[3] = (int)(((tensor_d1 >> 16) & 0xFFFFu) |
                ((tile_d0 & 0xFFFFu) << 16));                    // tile_dim0
  g1[4] = (int)(tile_d1 & 0xFFFFu);                              // tile_dim1 (tile_dim2=0)
  g1[5] = (int)(uint32_t)(row_stride_elems & 0xFFFFFFFFu);       // tensor_dim0_stride lo
  g1[6] = (int)(uint32_t)((row_stride_elems >> 32) & 0xFFFFu);
  g1[7] = 0;
  v4i_ gz = {0, 0, 0, 0};
#if defined(__clang_major__) && (__clang_major__ >= 23)
  v8i_ gz8 = {0, 0, 0, 0, 0, 0, 0, 0};
  __builtin_amdgcn_tensor_load_to_lds(g0, g1, gz, gz, gz8, 0);
#else
  __builtin_amdgcn_tensor_load_to_lds(g0, g1, gz, gz, 0);
#endif
}
#endif  // ATHENA_HAVE_TDM

// ---------------------------------------------------------------------------
// GEMM: Y[m,n] = sum_k X[m,k] * W[n,k] + bias[n]      (torch Linear: X @ W^T + b)
// Block tile 128x128, K tiles of 64, 8 waves each computing 32x64.
// OUT_MODE: 0 = f32 flat [M][D]     (output projection -> d_out)
//           1 = f16 heads [B*H][S][DH]      (Q, K)
//           2 = f16 heads transposed [B*H][DH][S]  (V; vectorized b128 stores)
// ---------------------------------------------------------------------------
template <bool IN_HALF, int OUT_MODE>
__global__ __launch_bounds__(256)
void gemm128_kernel(const void* __restrict__ Xv, const float* __restrict__ W,
                    const float* __restrict__ bias, void* __restrict__ Yv) {
  __shared__ _Float16 As[128][72];   // [m][k], 64 K + 8 pad (144B pitch)
  __shared__ _Float16 Bs[128][72];   // [n][k]

  const int tid  = threadIdx.x;
  const int lane = tid & 31;
  const int wave = tid >> 5;
  const int waveM = wave & 3;
  const int waveN = wave >> 2;
  const int bM = blockIdx.y * 128;
  const int bN = blockIdx.x * 128;

  v8f acc[2][4] = {};

  const int lrow = tid >> 1;         // 0..127
  const int lcb  = (tid & 1) * 32;   // 0 or 32

  for (int kk = 0; kk < D_; kk += 64) {
    // ---- stage A tile (128x64 f16) ----
    if constexpr (IN_HALF) {
#if ATHENA_HAVE_TDM
      if (wave == 0) {
        tdm_load_2d_f16((uint32_t)(uintptr_t)&As[0][0],
                        (const _Float16*)Xv + (size_t)bM * D_ + kk,
                        /*tile*/ 64, 128, /*tensor*/ 64, 128,
                        /*row stride*/ D_, /*pad +16B / 128B*/ 4, 3, 1);
      }
#else
      const _Float16* src = (const _Float16*)Xv + (size_t)(bM + lrow) * D_ + kk + lcb;
#pragma unroll
      for (int j = 0; j < 32; j += 8)
        *(v8h*)&As[lrow][lcb + j] = *(const v8h*)(src + j);
#endif
    } else {
      const float* src = (const float*)Xv + (size_t)(bM + lrow) * D_ + kk + lcb;
#pragma unroll
      for (int j = 0; j < 32; j += 8) {
        float4 f0 = *(const float4*)(src + j);
        float4 f1 = *(const float4*)(src + j + 4);
        v8h h;
        h[0] = (_Float16)f0.x; h[1] = (_Float16)f0.y;
        h[2] = (_Float16)f0.z; h[3] = (_Float16)f0.w;
        h[4] = (_Float16)f1.x; h[5] = (_Float16)f1.y;
        h[6] = (_Float16)f1.z; h[7] = (_Float16)f1.w;
        *(v8h*)&As[lrow][lcb + j] = h;
      }
      if (kk + 64 < D_) __builtin_prefetch(src + 64, 0, 1);
    }
    // ---- stage B tile: rows n of W (f32 -> f16) ----
    {
      const float* src = W + (size_t)(bN + lrow) * D_ + kk + lcb;
#pragma unroll
      for (int j = 0; j < 32; j += 8) {
        float4 f0 = *(const float4*)(src + j);
        float4 f1 = *(const float4*)(src + j + 4);
        v8h h;
        h[0] = (_Float16)f0.x; h[1] = (_Float16)f0.y;
        h[2] = (_Float16)f0.z; h[3] = (_Float16)f0.w;
        h[4] = (_Float16)f1.x; h[5] = (_Float16)f1.y;
        h[6] = (_Float16)f1.z; h[7] = (_Float16)f1.w;
        *(v8h*)&Bs[lrow][lcb + j] = h;
      }
      if (kk + 64 < D_) __builtin_prefetch(src + 64, 0, 1);
    }
#if ATHENA_HAVE_TDM
    if (IN_HALF && wave == 0) __builtin_amdgcn_s_wait_tensorcnt((short)0);
#endif
    __syncthreads();

#pragma unroll
    for (int ks = 0; ks < 2; ++ks) {
      v16h afr[2], bfr[4];
#pragma unroll
      for (int mi = 0; mi < 2; ++mi)
        afr[mi] = load_frag_a(&As[waveM * 32 + mi * 16][ks * 32], 72, lane);
#pragma unroll
      for (int ni = 0; ni < 4; ++ni)
        bfr[ni] = load_frag_b(&Bs[waveN * 64 + ni * 16][ks * 32], 72, lane);
#pragma unroll
      for (int mi = 0; mi < 2; ++mi)
#pragma unroll
        for (int ni = 0; ni < 4; ++ni)
          acc[mi][ni] = wmma_f16(afr[mi], bfr[ni], acc[mi][ni]);
    }
    __syncthreads();
  }

  // ---- epilogue: C/D layout = lane (l&15)=col, row = r + 8*(l>>4) ----
  const int hi = lane >> 4, lo = lane & 15;
#pragma unroll
  for (int mi = 0; mi < 2; ++mi) {
#pragma unroll
    for (int ni = 0; ni < 4; ++ni) {
      const int gn = bN + waveN * 64 + ni * 16 + lo;
      const float bval = bias[gn];
      if constexpr (OUT_MODE == 2) {
        // V^T per head: [B*H][DH][S]; 8 rows per lane are contiguous in S.
        v8h hv;
#pragma unroll
        for (int r = 0; r < 8; ++r) hv[r] = (_Float16)(acc[mi][ni][r] + bval);
        const int gm0 = bM + waveM * 32 + mi * 16 + hi * 8;
        const int b = gm0 >> 11, s0 = gm0 & (S_ - 1);
        const int h = gn >> 6,   dh = gn & (DH_ - 1);
        _Float16* Y = (_Float16*)Yv;
        *(v8h*)&Y[((size_t)(b * H_ + h) * DH_ + dh) * S_ + s0] = hv;
      } else {
#pragma unroll
        for (int r = 0; r < 8; ++r) {
          const int gm = bM + waveM * 32 + mi * 16 + r + hi * 8;
          const float v = acc[mi][ni][r] + bval;
          if constexpr (OUT_MODE == 1) {
            _Float16* Y = (_Float16*)Yv;
            const int b = gm >> 11, s = gm & (S_ - 1);
            const int h = gn >> 6,  dh = gn & (DH_ - 1);
            Y[((size_t)(b * H_ + h) * S_ + s) * DH_ + dh] = (_Float16)v;
          } else {
            float* Y = (float*)Yv;
            Y[(size_t)gm * D_ + gn] = v;
          }
        }
      }
    }
  }
}

// ---------------------------------------------------------------------------
// Flash attention: grid (S/128, B*H), 256 threads (8 waves).
// Wave w owns 16 query rows; streams 64-key tiles with online softmax.
// Double-buffered pipeline: TDM DMA for tile t+1 is issued (waves 0/1) before
// computing on tile t; TENSORcnt drained only at the end of the iteration, so
// the DMA runs entirely in the shadow of the WMMA/softmax work.
// ---------------------------------------------------------------------------
__global__ __launch_bounds__(256)
void flash_attn_kernel(const _Float16* __restrict__ Q, const _Float16* __restrict__ K,
                       const _Float16* __restrict__ V, _Float16* __restrict__ Ctx) {
  __shared__ _Float16 Ks[2][64][72];     // [buf][key][dh]
  __shared__ _Float16 Vt[2][64][72];     // [buf][dh][key]
  __shared__ _Float16 Ps[8][16][72];     // per-wave P tile [qrow][key]

  const int tid  = threadIdx.x;
  const int lane = tid & 31;
  const int wave = tid >> 5;
  const int hi = lane >> 4, lo = lane & 15;

  const int bh = blockIdx.y;
  const int q0 = blockIdx.x * 128 + wave * 16;

  const _Float16* Qb = Q + (size_t)bh * S_ * DH_;   // [S][DH]
  const _Float16* Kb = K + (size_t)bh * S_ * DH_;   // [S][DH]
  const _Float16* Vb = V + (size_t)bh * DH_ * S_;   // [DH][S]  (transposed)

  const int trow = tid >> 2;          // 0..63
  const int tcb  = (tid & 3) * 16;    // 0,16,32,48

  // ---- tile staging (TDM: waves 0/1 issue K and V^T DMAs in parallel) ----
  auto stage_tiles = [&](int kt, int buf) {
#if ATHENA_HAVE_TDM
    if (wave == 0) {
      tdm_load_2d_f16((uint32_t)(uintptr_t)&Ks[buf][0][0], Kb + (size_t)kt * DH_,
                      /*tile*/ 64, 64, /*tensor*/ 64, 64,
                      /*row stride*/ DH_, /*pad +16B / 128B*/ 4, 3, 1);
    } else if (wave == 1) {
      tdm_load_2d_f16((uint32_t)(uintptr_t)&Vt[buf][0][0], Vb + kt,
                      /*tile*/ 64, 64, /*tensor*/ 64, 64,
                      /*row stride*/ S_, /*pad +16B / 128B*/ 4, 3, 1);
    }
#else
    {
      const _Float16* src = Kb + (size_t)(kt + trow) * DH_ + tcb;
      v8h k0 = *(const v8h*)src, k1 = *(const v8h*)(src + 8);
      *(v8h*)&Ks[buf][trow][tcb]     = k0;
      *(v8h*)&Ks[buf][trow][tcb + 8] = k1;
    }
    {
      const _Float16* src = Vb + (size_t)trow * S_ + kt + tcb;
      v8h p0 = *(const v8h*)src, p1 = *(const v8h*)(src + 8);
      *(v8h*)&Vt[buf][trow][tcb]     = p0;
      *(v8h*)&Vt[buf][trow][tcb + 8] = p1;
    }
    if (kt + 64 < S_) {
      __builtin_prefetch(Kb + (size_t)(kt + 64 + trow) * DH_ + tcb, 0, 1);
      __builtin_prefetch(Vb + (size_t)trow * S_ + kt + 64 + tcb, 0, 1);
    }
#endif
  };

  // Q A-fragments, loaded once straight from global
  v16h aQ[2];
  {
    const _Float16* qr = Qb + (size_t)(q0 + lo) * DH_;
#pragma unroll
    for (int ks = 0; ks < 2; ++ks) {
      v8h l0 = *(const v8h*)(qr + ks * 32 + hi * 8);
      v8h h0 = *(const v8h*)(qr + ks * 32 + 16 + hi * 8);
      aQ[ks] = cat16(l0, h0);
    }
  }

  float m_i[8], l_i[8];
  v8f oacc[4] = {};
#pragma unroll
  for (int r = 0; r < 8; ++r) { m_i[r] = -1e30f; l_i[r] = 0.0f; }

  // ---- prologue: stage tile 0 ----
  stage_tiles(0, 0);
#if ATHENA_HAVE_TDM
  if (wave < 2) __builtin_amdgcn_s_wait_tensorcnt((short)0);
#endif
  __syncthreads();

  for (int kt = 0; kt < S_; kt += 64) {
    const int cur = (kt >> 6) & 1;
    const int nxt = cur ^ 1;

    // ---- kick off DMA for the NEXT tile into the alternate buffer ----
    if (kt + 64 < S_) stage_tiles(kt + 64, nxt);

    // ---- S = Q @ K^T : 16 x 64 scores per wave ----
    v8f sc[4] = {};
#pragma unroll
    for (int ks = 0; ks < 2; ++ks) {
#pragma unroll
      for (int ni = 0; ni < 4; ++ni) {
        v16h bf = load_frag_b(&Ks[cur][ni * 16][ks * 32], 72, lane);
        sc[ni] = wmma_f16(aQ[ks], bf, sc[ni]);
      }
    }
#pragma unroll
    for (int ni = 0; ni < 4; ++ni)
#pragma unroll
      for (int r = 0; r < 8; ++r) sc[ni][r] *= 0.125f;   // 1/sqrt(64)

    // ---- online softmax (lane lo = col, vgpr r -> row r + 8*hi) ----
    float corr[8];
#pragma unroll
    for (int r = 0; r < 8; ++r) {
      const float rowm =
          red16_max(fmaxf(fmaxf(sc[0][r], sc[1][r]), fmaxf(sc[2][r], sc[3][r])));
      const float mnew = fmaxf(m_i[r], rowm);
      corr[r] = __expf(m_i[r] - mnew);
      m_i[r]  = mnew;
    }
#pragma unroll
    for (int ni = 0; ni < 4; ++ni)
#pragma unroll
      for (int r = 0; r < 8; ++r) sc[ni][r] = __expf(sc[ni][r] - m_i[r]);
#pragma unroll
    for (int r = 0; r < 8; ++r) {
      const float rs = red16_sum(sc[0][r] + sc[1][r] + sc[2][r] + sc[3][r]);
      l_i[r] = l_i[r] * corr[r] + rs;
#pragma unroll
      for (int ni = 0; ni < 4; ++ni) oacc[ni][r] *= corr[r];
    }

    // ---- P round-trip through per-wave LDS to become an A fragment ----
#pragma unroll
    for (int ni = 0; ni < 4; ++ni)
#pragma unroll
      for (int r = 0; r < 8; ++r)
        Ps[wave][r + hi * 8][ni * 16 + lo] = (_Float16)sc[ni][r];

    asm volatile("s_wait_dscnt 0" ::: "memory");  // same-wave cross-lane LDS dep

    // ---- O += P @ V ----
#pragma unroll
    for (int ks = 0; ks < 2; ++ks) {
      v16h aP = load_frag_a(&Ps[wave][0][ks * 32], 72, lane);
#pragma unroll
      for (int ni = 0; ni < 4; ++ni) {
        v16h bf = load_frag_b(&Vt[cur][ni * 16][ks * 32], 72, lane);
        oacc[ni] = wmma_f16(aP, bf, oacc[ni]);
      }
    }

    // ---- drain next-tile DMA (ran in the shadow of the compute above) ----
#if ATHENA_HAVE_TDM
    if (kt + 64 < S_ && wave < 2) __builtin_amdgcn_s_wait_tensorcnt((short)0);
#endif
    __syncthreads();
  }

  // ---- normalize and store ctx as [B][S][D] f16 ----
  const int bb = bh >> 4, h = bh & (H_ - 1);
#pragma unroll
  for (int ni = 0; ni < 4; ++ni)
#pragma unroll
    for (int r = 0; r < 8; ++r) {
      const int s  = q0 + r + hi * 8;
      const int dh = ni * 16 + lo;
      const float val = oacc[ni][r] / l_i[r];
      Ctx[((size_t)bb * S_ + s) * D_ + h * DH_ + dh] = (_Float16)val;
    }
}

// ---------------------------------------------------------------------------
extern "C" void kernel_launch(void* const* d_in, const int* in_sizes, int n_in,
                              void* d_out, int out_size, void* d_ws, size_t ws_size,
                              hipStream_t stream) {
  (void)in_sizes; (void)n_in; (void)out_size; (void)ws_size;

  const float* q  = (const float*)d_in[0];
  const float* k  = (const float*)d_in[1];
  const float* v  = (const float*)d_in[2];
  const float* Wq = (const float*)d_in[3];
  const float* bq = (const float*)d_in[4];
  const float* Wk = (const float*)d_in[5];
  const float* bk = (const float*)d_in[6];
  const float* Wv = (const float*)d_in[7];
  const float* bv = (const float*)d_in[8];
  const float* Wo = (const float*)d_in[9];
  const float* bo = (const float*)d_in[10];

  const size_t NH = (size_t)B_ * S_ * D_;       // 8,388,608 elements
  _Float16* Qh  = (_Float16*)d_ws;              // [B*H][S][DH] f16, 16 MB
  _Float16* Kh  = Qh + NH;                      // [B*H][S][DH] f16, 16 MB
  _Float16* Vh  = Kh + NH;                      // [B*H][DH][S] f16 (transposed), 16 MB
  _Float16* Ctx = Vh + NH;                      // [B][S][D] f16, 16 MB

  dim3 blk(256);
  dim3 gp(D_ / 128, (B_ * S_) / 128);           // (8, 64)

  gemm128_kernel<false, 1><<<gp, blk, 0, stream>>>(q, Wq, bq, Qh);
  gemm128_kernel<false, 1><<<gp, blk, 0, stream>>>(k, Wk, bk, Kh);
  gemm128_kernel<false, 2><<<gp, blk, 0, stream>>>(v, Wv, bv, Vh);

  flash_attn_kernel<<<dim3(S_ / 128, B_ * H_), blk, 0, stream>>>(Qh, Kh, Vh, Ctx);

  gemm128_kernel<true, 0><<<gp, blk, 0, stream>>>(Ctx, Wo, bo, (float*)d_out);
}
